// MultiHeadLatentAttention_63161789055663
// MI455X (gfx1250) — matrix operand order repaired and checked by reading, not compiled, not run on hardware
//
#include <hip/hip_runtime.h>

// ---------------------------------------------------------------------------
// MLA forward for MI455X (gfx1250): bf16 WMMA GEMMs (async-LDS double-buffered)
// + flash attention with online softmax.
// ---------------------------------------------------------------------------

typedef __attribute__((ext_vector_type(16))) __bf16 v16bf;
typedef __attribute__((ext_vector_type(8)))  float  v8f;
typedef int vi4 __attribute__((vector_size(16)));   // pointee type of async-LDS builtin

#define AS_GLOBAL __attribute__((address_space(1)))
#define AS_LDS    __attribute__((address_space(3)))

#define D_MODEL   2048
#define NUM_HEADS 16
#define D_HEAD    128
#define D_LATENT  512
#define BATCH     2
#define SEQ       2048
#define M_TOKENS  (BATCH * SEQ)   // 4096

// gfx1250 async global->LDS copy path (ASYNCcnt-tracked).  Guarded so the file
// compiles on toolchains that don't expose the builtins (falls back to
// synchronous register-bounce staging).
#if defined(__has_builtin)
#if __has_builtin(__builtin_amdgcn_global_load_async_to_lds_b128) && \
    __has_builtin(__builtin_amdgcn_s_wait_asynccnt)
#define USE_ASYNC_LDS 1
#endif
#endif

__device__ __forceinline__ unsigned short f32_to_bf16(float f) {
  unsigned int u = __builtin_bit_cast(unsigned int, f);
  u += 0x7FFFu + ((u >> 16) & 1u);   // round-to-nearest-even
  return (unsigned short)(u >> 16);
}

union Frag {
  v16bf v;
  uint4 q[2];
  unsigned short u[16];
};

// A operand 16x32 bf16: lane l -> row (l&15); K chunks {c0..c0+7, 16+c0..+7},
// c0 = 8*(l>=16).  Two 16B loads.
__device__ __forceinline__ void load_fragA(Frag& f, const unsigned short* base,
                                           int ld, int row, int k0, int lane) {
  const int r  = row + (lane & 15);
  const int c0 = (lane >> 4) << 3;   // 0 or 8
  const unsigned short* p = base + (size_t)r * ld + k0 + c0;
  f.q[0] = *(const uint4*)(p);
  f.q[1] = *(const uint4*)(p + 16);
}

// B operand 32x16 bf16 from a [N,K] row-major buffer (i.e. B^T): lane l ->
// column (l&15); K chunks {c0..c0+7, 16+c0..+7}, c0 = 8*(l>=16).  Two 16B loads.
__device__ __forceinline__ void load_fragB(Frag& f, const unsigned short* base,
                                           int ld, int col, int k0, int lane) {
  const int n    = col + (lane & 15);
  const int koff = (lane >> 4) << 3; // 0 or 8
  const unsigned short* p = base + (size_t)n * ld + k0 + koff;
  f.q[0] = *(const uint4*)(p);
  f.q[1] = *(const uint4*)(p + 16);
}

// Same layouts reading a staged LDS tile with padded row stride (elements).
__device__ __forceinline__ void load_fragA_lds(Frag& f, const unsigned short* s,
                                               int stride, int row, int lane) {
  const int r  = row + (lane & 15);
  const int c0 = (lane >> 4) << 3;
  const unsigned short* p = s + r * stride + c0;
  f.q[0] = *(const uint4*)(p);
  f.q[1] = *(const uint4*)(p + 16);
}
__device__ __forceinline__ void load_fragB_lds(Frag& f, const unsigned short* s,
                                               int stride, int col, int lane) {
  const int n    = col + (lane & 15);
  const int koff = (lane >> 4) << 3;
  const unsigned short* p = s + n * stride + koff;
  f.q[0] = *(const uint4*)(p);
  f.q[1] = *(const uint4*)(p + 16);
}

__device__ __forceinline__ v8f wmma_bf16(const Frag& a, const Frag& b, v8f c) {
  return __builtin_amdgcn_wmma_f32_16x16x32_bf16(false, a.v, false, b.v,
                                                 (short)0, c, false, false);
}

// ---------------------------------------------------------------------------
// Elementwise conversion / transpose kernels
// ---------------------------------------------------------------------------

__global__ void cvt_f32_to_bf16_kernel(const float* __restrict__ in,
                                       unsigned short* __restrict__ out, size_t n) {
  size_t i = (size_t)blockIdx.x * blockDim.x + threadIdx.x;
  if (i < n) out[i] = f32_to_bf16(in[i]);
}

// W[K,N] f32 -> Wt[N,K] bf16
__global__ void transpose_w_kernel(const float* __restrict__ W,
                                   unsigned short* __restrict__ Wt, int K, int N) {
  size_t i = (size_t)blockIdx.x * blockDim.x + threadIdx.x;
  if (i < (size_t)K * N) {
    int k = (int)(i / N);
    int n = (int)(i % N);
    Wt[(size_t)n * K + k] = f32_to_bf16(W[i]);
  }
}

// V bf16 [B*S, H*Dh] -> Vt bf16 [(b,h,d), S]
__global__ void transpose_v_kernel(const unsigned short* __restrict__ V,
                                   unsigned short* __restrict__ Vt) {
  size_t i = (size_t)blockIdx.x * blockDim.x + threadIdx.x;
  if (i < (size_t)M_TOKENS * D_MODEL) {
    int row = (int)(i / D_MODEL);
    int c   = (int)(i % D_MODEL);
    int b = row / SEQ, s = row % SEQ;
    int h = c / D_HEAD, d = c % D_HEAD;
    Vt[((size_t)(b * NUM_HEADS + h) * D_HEAD + d) * SEQ + s] = V[i];
  }
}

// ---------------------------------------------------------------------------
// bf16 WMMA GEMM: C[M,N] = A[M,K] @ Bt[N,K]^T + bias[N]
// Block = 256 threads = 8 waves (4M x 2N); wave tile 32x64; block tile 128x128.
// K streamed in 32-wide stages, double-buffered in LDS with async copies.
// LDS rows padded to 40 elements (80B, 16B-aligned, bank-conflict friendly).
// ---------------------------------------------------------------------------

#define TPAD 40  // padded row stride in elements for a 32-wide K stage

// Stage a rows x 32 bf16 tile into LDS (padded rows).  256 threads, each moves
// 16B units.
__device__ __forceinline__ void stage_tile(const unsigned short* __restrict__ g,
                                           int ldg, unsigned short* s,
                                           int rows, int tid) {
  const int units = rows * 4;  // 4 x 8-element units per row
  for (int u = tid; u < units; u += 256) {
    const int r = u >> 2;
    const int c = (u & 3) << 3;
#if defined(USE_ASYNC_LDS)
    __builtin_amdgcn_global_load_async_to_lds_b128(
        (AS_GLOBAL vi4*)(uintptr_t)(g + (size_t)r * ldg + c),
        (AS_LDS vi4*)(unsigned)(uintptr_t)(s + r * TPAD + c),
        0, 0);
#else
    *(uint4*)(s + r * TPAD + c) = *(const uint4*)(g + (size_t)r * ldg + c);
#endif
  }
}

__device__ __forceinline__ void wait_stage() {
#if defined(USE_ASYNC_LDS)
  __builtin_amdgcn_s_wait_asynccnt(0);
#endif
}

template <bool OUT_BF16>
__global__ __launch_bounds__(256) void gemm_bf16_kernel(
    const unsigned short* __restrict__ A, const unsigned short* __restrict__ Bt,
    const float* __restrict__ bias, void* __restrict__ Cout,
    int M, int N, int K) {
  __shared__ unsigned short sA[2][128 * TPAD];
  __shared__ unsigned short sB[2][128 * TPAD];

  const int tid  = threadIdx.x;
  const int lane = tid & 31;
  const int wid  = tid >> 5;
  const int wm   = wid >> 1;                 // 0..3
  const int wn   = wid & 1;                  // 0..1
  const int row0 = blockIdx.y * 128;
  const int col0 = blockIdx.x * 128;

  v8f acc[2][4];
#pragma unroll
  for (int i = 0; i < 2; ++i)
#pragma unroll
    for (int j = 0; j < 4; ++j)
#pragma unroll
      for (int r = 0; r < 8; ++r) acc[i][j][r] = 0.0f;

  // prologue: stage K-block 0 into buffer 0
  stage_tile(A  + (size_t)row0 * K, K, sA[0], 128, tid);
  stage_tile(Bt + (size_t)col0 * K, K, sB[0], 128, tid);

  int buf = 0;
  for (int k0 = 0; k0 < K; k0 += 32) {
    wait_stage();        // each wave waits for ITS async copies...
    __syncthreads();     // ...then the barrier makes the whole tile visible

    if (k0 + 32 < K) {   // overlap next stage with compute on current one
      stage_tile(A  + (size_t)row0 * K + k0 + 32, K, sA[buf ^ 1], 128, tid);
      stage_tile(Bt + (size_t)col0 * K + k0 + 32, K, sB[buf ^ 1], 128, tid);
    }

    Frag af[2], bf[4];
#pragma unroll
    for (int i = 0; i < 2; ++i)
      load_fragA_lds(af[i], sA[buf], TPAD, wm * 32 + i * 16, lane);
#pragma unroll
    for (int j = 0; j < 4; ++j)
      load_fragB_lds(bf[j], sB[buf], TPAD, wn * 64 + j * 16, lane);
#pragma unroll
    for (int i = 0; i < 2; ++i)
#pragma unroll
      for (int j = 0; j < 4; ++j)
        acc[i][j] = wmma_bf16(af[i], bf[j], acc[i][j]);

    __syncthreads();     // tile reads done before stage k+2 overwrites buf
    buf ^= 1;
  }

  const int nl = lane & 15;
  const int mb = (lane >> 4) * 8;
#pragma unroll
  for (int i = 0; i < 2; ++i) {
#pragma unroll
    for (int j = 0; j < 4; ++j) {
      const int ncol = col0 + wn * 64 + j * 16 + nl;
      const float bv = bias[ncol];
#pragma unroll
      for (int r = 0; r < 8; ++r) {
        const float vv  = acc[i][j][r] + bv;
        const size_t ix = (size_t)(row0 + wm * 32 + i * 16 + mb + r) * N + ncol;
        if (OUT_BF16) ((unsigned short*)Cout)[ix] = f32_to_bf16(vv);
        else          ((float*)Cout)[ix]          = vv;
      }
    }
  }
}

// ---------------------------------------------------------------------------
// Flash attention: one wave per (b, h, 16-query tile).  Keys streamed in
// 32-wide blocks with online softmax.  4 waves/block share one qb so loop
// trip counts match.  P tile transposed C-layout -> A-layout through LDS.
// ---------------------------------------------------------------------------

__global__ __launch_bounds__(128) void mla_flash_kernel(
    const unsigned short* __restrict__ Qb, const unsigned short* __restrict__ Kb,
    const unsigned short* __restrict__ Vt, unsigned short* __restrict__ Ob) {
  __shared__ float sP[4][16 * 32];

  const int lane = threadIdx.x & 31;
  const int wid  = threadIdx.x >> 5;
  const int bx   = blockIdx.x;
  const int qb = bx & (SEQ / 16 - 1);        // 0..127
  const int hg = (bx >> 7) & 3;
  const int b  = bx >> 9;
  const int h  = (hg << 2) + wid;            // 4 heads per block, same qb

  const int nl   = lane & 15;
  const int half = lane >> 4;
  const int row0 = b * SEQ + qb * 16;

  Frag qf[4];
#pragma unroll
  for (int ks = 0; ks < 4; ++ks)
    load_fragA(qf[ks], Qb, D_MODEL, row0, h * D_HEAD + ks * 32, lane);

  v8f acc[8];
  float rm[8], rs[8];
#pragma unroll
  for (int t = 0; t < 8; ++t) {
#pragma unroll
    for (int r = 0; r < 8; ++r) acc[t][r] = 0.0f;
    rm[t] = -INFINITY;
    rs[t] = 0.0f;
  }

  const float scale = 0.08838834764831845f;  // 1/sqrt(128)
  const int nkb = (qb * 16 + 16 + 31) >> 5;  // causal: key blocks needed

  for (int kb = 0; kb < nkb; ++kb) {
    const int kbase = kb * 32;

    // ---- S = Q @ K^T (two 16-key N tiles, K-dim = 128 in 4 steps) ----
    v8f st[2];
#pragma unroll
    for (int t = 0; t < 2; ++t) {
#pragma unroll
      for (int r = 0; r < 8; ++r) st[t][r] = 0.0f;
#pragma unroll
      for (int ks = 0; ks < 4; ++ks) {
        Frag kf;
        load_fragB(kf, Kb, D_MODEL, b * SEQ + kbase + t * 16,
                   h * D_HEAD + ks * 32, lane);
        st[t] = wmma_bf16(qf[ks], kf, st[t]);
      }
    }

    // ---- scale + causal mask + per-row block max ----
    float bm[8];
#pragma unroll
    for (int r = 0; r < 8; ++r) {
      const int qg = qb * 16 + r + 8 * half;
      float s0 = st[0][r] * scale;
      float s1 = st[1][r] * scale;
      if (kbase + nl > qg)      s0 = -INFINITY;
      if (kbase + 16 + nl > qg) s1 = -INFINITY;
      st[0][r] = s0;
      st[1][r] = s1;
      float m = fmaxf(s0, s1);
#pragma unroll
      for (int off = 1; off < 16; off <<= 1)
        m = fmaxf(m, __shfl_xor(m, off, 32));
      bm[r] = m;
    }

    // ---- online softmax update ----
    float nm[8], alpha[8];
#pragma unroll
    for (int r = 0; r < 8; ++r) {
      nm[r]    = fmaxf(rm[r], bm[r]);
      alpha[r] = __expf(rm[r] - nm[r]);
      rm[r]    = nm[r];
    }
#pragma unroll
    for (int r = 0; r < 8; ++r) {
      const float p0 = __expf(st[0][r] - nm[r]);
      const float p1 = __expf(st[1][r] - nm[r]);
      st[0][r] = p0;
      st[1][r] = p1;
      float s = p0 + p1;
#pragma unroll
      for (int off = 1; off < 16; off <<= 1) s += __shfl_xor(s, off, 32);
      rs[r] = rs[r] * alpha[r] + s;
    }
#pragma unroll
    for (int dt = 0; dt < 8; ++dt)
#pragma unroll
      for (int r = 0; r < 8; ++r) acc[dt][r] *= alpha[r];

    // ---- P: C-layout -> A-layout via per-wave LDS ----
#pragma unroll
    for (int r = 0; r < 8; ++r) {
      const int m = r + 8 * half;
      sP[wid][m * 32 + nl]      = st[0][r];
      sP[wid][m * 32 + 16 + nl] = st[1][r];
    }
    Frag pf;
    {
      const int prow = nl;
      const int c0   = half * 8;
#pragma unroll
      for (int e = 0; e < 8; ++e)
        pf.u[e] = f32_to_bf16(sP[wid][prow * 32 + c0 + e]);
#pragma unroll
      for (int e = 0; e < 8; ++e)
        pf.u[8 + e] = f32_to_bf16(sP[wid][prow * 32 + 16 + c0 + e]);
    }

    // ---- O += P @ V  (B operand streamed from head-transposed V) ----
#pragma unroll
    for (int dt = 0; dt < 8; ++dt) {
      Frag vf;
      load_fragB(vf, Vt, SEQ, (b * NUM_HEADS + h) * D_HEAD + dt * 16, kbase, lane);
      acc[dt] = wmma_bf16(pf, vf, acc[dt]);
    }
  }

  // ---- epilogue: normalize and store bf16 ----
  float inv[8];
#pragma unroll
  for (int r = 0; r < 8; ++r) inv[r] = 1.0f / rs[r];
#pragma unroll
  for (int dt = 0; dt < 8; ++dt) {
#pragma unroll
    for (int r = 0; r < 8; ++r) {
      const size_t orow = (size_t)row0 + r + 8 * half;
      Ob[orow * D_MODEL + h * D_HEAD + dt * 16 + nl] =
          f32_to_bf16(acc[dt][r] * inv[r]);
    }
  }
}

// ---------------------------------------------------------------------------
// Launch
// ---------------------------------------------------------------------------

extern "C" void kernel_launch(void* const* d_in, const int* in_sizes, int n_in,
                              void* d_out, int out_size, void* d_ws, size_t ws_size,
                              hipStream_t stream) {
  (void)in_sizes; (void)n_in; (void)out_size; (void)ws_size;
  const float* x   = (const float*)d_in[0];
  // d_in[1] is the causal mask; it is tril, so masking is computed analytically.
  const float* Wq  = (const float*)d_in[2];
  const float* bq  = (const float*)d_in[3];
  const float* Wkv = (const float*)d_in[4];
  const float* bkv = (const float*)d_in[5];
  const float* Wk  = (const float*)d_in[6];
  const float* bk  = (const float*)d_in[7];
  const float* Wv  = (const float*)d_in[8];
  const float* bv  = (const float*)d_in[9];
  const float* Wo  = (const float*)d_in[10];
  const float* bo  = (const float*)d_in[11];

  char* ws = (char*)d_ws;
  size_t off = 0;
  auto alloc = [&](size_t elems) -> unsigned short* {
    unsigned short* p = (unsigned short*)(ws + off);
    off += ((elems * sizeof(unsigned short) + 255) / 256) * 256;
    return p;
  };
  unsigned short* Xb   = alloc((size_t)M_TOKENS * D_MODEL);
  unsigned short* WqT  = alloc((size_t)D_MODEL * D_MODEL);
  unsigned short* WkvT = alloc((size_t)D_LATENT * D_MODEL);
  unsigned short* WkT  = alloc((size_t)D_MODEL * D_LATENT);
  unsigned short* WvT  = alloc((size_t)D_MODEL * D_LATENT);
  unsigned short* WoT  = alloc((size_t)D_MODEL * D_MODEL);
  unsigned short* Qbuf = alloc((size_t)M_TOKENS * D_MODEL);
  unsigned short* Lbuf = alloc((size_t)M_TOKENS * D_LATENT);
  unsigned short* Kbuf = alloc((size_t)M_TOKENS * D_MODEL);
  unsigned short* Vbuf = alloc((size_t)M_TOKENS * D_MODEL);
  unsigned short* Vt   = alloc((size_t)M_TOKENS * D_MODEL);
  unsigned short* Obuf = alloc((size_t)M_TOKENS * D_MODEL);

  const size_t nx = (size_t)M_TOKENS * D_MODEL;
  cvt_f32_to_bf16_kernel<<<(unsigned)((nx + 255) / 256), 256, 0, stream>>>(x, Xb, nx);

  auto nblk = [](size_t n) { return (unsigned)((n + 255) / 256); };
  transpose_w_kernel<<<nblk((size_t)D_MODEL * D_MODEL), 256, 0, stream>>>(Wq,  WqT,  D_MODEL,  D_MODEL);
  transpose_w_kernel<<<nblk((size_t)D_MODEL * D_LATENT), 256, 0, stream>>>(Wkv, WkvT, D_MODEL,  D_LATENT);
  transpose_w_kernel<<<nblk((size_t)D_LATENT * D_MODEL), 256, 0, stream>>>(Wk,  WkT,  D_LATENT, D_MODEL);
  transpose_w_kernel<<<nblk((size_t)D_LATENT * D_MODEL), 256, 0, stream>>>(Wv,  WvT,  D_LATENT, D_MODEL);
  transpose_w_kernel<<<nblk((size_t)D_MODEL * D_MODEL), 256, 0, stream>>>(Wo,  WoT,  D_MODEL,  D_MODEL);

  dim3 gFull(D_MODEL / 128, M_TOKENS / 128);   // N=2048 GEMMs: (16, 32)
  dim3 gLat (D_LATENT / 128, M_TOKENS / 128);  // N=512 GEMM:   (4, 32)

  // Q = X @ Wq + bq                    [4096 x 2048], K=2048
  gemm_bf16_kernel<true><<<gFull, 256, 0, stream>>>(Xb, WqT, bq, Qbuf,
                                                    M_TOKENS, D_MODEL, D_MODEL);
  // latent = X @ Wkv + bkv             [4096 x 512],  K=2048
  gemm_bf16_kernel<true><<<gLat, 256, 0, stream>>>(Xb, WkvT, bkv, Lbuf,
                                                   M_TOKENS, D_LATENT, D_MODEL);
  // K = latent @ Wk + bk               [4096 x 2048], K=512
  gemm_bf16_kernel<true><<<gFull, 256, 0, stream>>>(Lbuf, WkT, bk, Kbuf,
                                                    M_TOKENS, D_MODEL, D_LATENT);
  // V = latent @ Wv + bv               [4096 x 2048], K=512
  gemm_bf16_kernel<true><<<gFull, 256, 0, stream>>>(Lbuf, WvT, bv, Vbuf,
                                                    M_TOKENS, D_MODEL, D_LATENT);

  transpose_v_kernel<<<nblk(nx), 256, 0, stream>>>(Vbuf, Vt);

  // flash attention: B * (H/4) * (S/16) blocks of 4 waves
  mla_flash_kernel<<<BATCH * (NUM_HEADS / 4) * (SEQ / 16), 128, 0, stream>>>(
      Qbuf, Kbuf, Vt, Obuf);

  // out = O @ Wo + bo  (fp32 output)
  gemm_bf16_kernel<false><<<gFull, 256, 0, stream>>>(Obuf, WoT, bo, (float*)d_out,
                                                     M_TOKENS, D_MODEL, D_MODEL);
}
